// SUBNET_66099546685449
// MI455X (gfx1250) — matrix operand order repaired and checked by reading, not compiled, not run on hardware
//
#include <hip/hip_runtime.h>

#define BATCH 8192
#define TT    128

typedef __attribute__((ext_vector_type(16))) _Float16 v16h;
typedef __attribute__((ext_vector_type(8)))  _Float16 v8h;
typedef __attribute__((ext_vector_type(8)))  float    v8f;

static __device__ inline v8f zero8f() {
  v8f c;
#pragma unroll
  for (int i = 0; i < 8; ++i) c[i] = 0.f;
  return c;
}

static __device__ inline v8h zero8h() {
  v8h c;
#pragma unroll
  for (int i = 0; i < 8; ++i) c[i] = (_Float16)0.f;
  return c;
}

static __device__ inline v16h shuf16(v8h lo, v8h hi) {
  return __builtin_shufflevector(lo, hi, 0,1,2,3,4,5,6,7,8,9,10,11,12,13,14,15);
}

// D = A(16x32 f16) x B(32x16 f16) + C, f32 accumulate
static __device__ inline v8f wmma16(v16h a, v16h b, v8f c) {
  return __builtin_amdgcn_wmma_f32_16x16x32_f16(false, a, false, b, (short)0, c,
                                                false, false);
}

// B operand: packed as [tile][lane][16 halfs] -> one contiguous 32B load per lane
static __device__ inline v16h loadB(const _Float16* base, int tile, int lane) {
  return *(const v16h*)(base + tile * 512 + lane * 16);
}

// A operand from a 16x64 f16 row-major LDS tile (stride 64 halfs), K-tile kt.
// lane<16: row=lane,   elems = K{kt*32+0..7}, K{kt*32+16..23}
// lane>=16: row=lane-16, elems = K{kt*32+8..15}, K{kt*32+24..31}
static __device__ inline v16h loadA64(const _Float16* buf, int kt, int lane) {
  const int row = lane & 15;
  const int s8  = (lane >> 4) << 3;
  const _Float16* p = buf + row * 64 + kt * 32 + s8;
  v8h lo = *(const v8h*)p;
  v8h hi = *(const v8h*)(p + 16);
  return shuf16(lo, hi);
}

// Store 16x16 f32 D tile (C/D layout) into 16x64 row-major f16 with bias+relu.
static __device__ inline void store_relu(_Float16* buf, v8f c, float bias, int lane) {
  const int n  = lane & 15;
  const int mb = (lane >> 4) << 3;
#pragma unroll
  for (int v = 0; v < 8; ++v) {
    float x = c[v] + bias;
    buf[(mb + v) * 64 + n] = (_Float16)(x > 0.f ? x : 0.f);
  }
}

static __device__ inline v8h cvt8(const float* p) {
  v8h r;
#pragma unroll
  for (int i = 0; i < 8; ++i) r[i] = (_Float16)p[i];
  return r;
}

// ---------------------------------------------------------------------------
// Pack f32 weight W[din x dout] (row-major) into WMMA-B per-lane layout:
// tile = kt*ntiles + nt ; element (lane,e): N = nt*16 + (lane&15),
// K = kt*32 + (e<8 ? e : e+8) + (lane<16 ? 0 : 8). Zero-padded.
// ---------------------------------------------------------------------------
__global__ void pack_kernel(const float* __restrict__ W, _Float16* __restrict__ dst,
                            int din, int dout, int ntiles) {
  const int tile = blockIdx.x;
  const int nt   = tile % ntiles;
  const int kt   = tile / ntiles;
  const int i    = threadIdx.x;      // 0..511
  const int lane = i >> 4;
  const int e    = i & 15;
  const int n    = nt * 16 + (lane & 15);
  const int k    = kt * 32 + ((e < 8) ? e : (e + 8)) + ((lane < 16) ? 0 : 8);
  float v = (k < din && n < dout) ? W[k * dout + n] : 0.f;
  dst[tile * 512 + i] = (_Float16)v;
}

// ---------------------------------------------------------------------------
// Encoder: x0 = MLP_res(concat(upast, ypast)), din=80 (pad 96), out 16.
// One wave per 16-row tile, 4 waves/block.
// Packed-tile map (global): eW1 @0 (3x4), eW2 @12 (2x4), eW3 @20 (2x1), eWr @22 (3x1)
// ---------------------------------------------------------------------------
__global__ __launch_bounds__(128) void encoder_kernel(
    const float* __restrict__ upast, const float* __restrict__ ypast,
    const _Float16* __restrict__ wpack,
    const float* __restrict__ eb1, const float* __restrict__ eb2,
    const float* __restrict__ eb3, const float* __restrict__ ebr,
    _Float16* __restrict__ x0g) {
  __shared__ __align__(16) _Float16 hb[4][16 * 64];
  const int tid = threadIdx.x, wave = tid >> 5, lane = tid & 31;
  const int b0 = (blockIdx.x * 4 + wave) * 16;
  const int n = lane & 15, s8 = (lane >> 4) << 3;
  const bool lo = lane < 16;

  const float* up = upast + (size_t)(b0 + n) * 40;
  const float* yp = ypast + (size_t)(b0 + n) * 40;
  // z = [upast(40), ypast(40), pad(16)] ; build 3 A K-tiles (all 8-float segs)
  v16h A0 = shuf16(cvt8(lo ? up + 0  : up + 8),  cvt8(lo ? up + 16 : up + 24));
  v16h A1 = shuf16(cvt8(lo ? up + 32 : yp + 0),  cvt8(lo ? yp + 8  : yp + 16));
  v16h A2 = shuf16(cvt8(lo ? yp + 24 : yp + 32), zero8h());

  const _Float16* eW1 = wpack + 0  * 512;
  const _Float16* eW2 = wpack + 12 * 512;
  const _Float16* eW3 = wpack + 20 * 512;
  const _Float16* eWr = wpack + 22 * 512;
  _Float16* H = hb[wave];

#pragma unroll
  for (int nt = 0; nt < 4; ++nt) {
    v8f c = zero8f();
    c = wmma16(A0, loadB(eW1, 0 * 4 + nt, lane), c);
    c = wmma16(A1, loadB(eW1, 1 * 4 + nt, lane), c);
    c = wmma16(A2, loadB(eW1, 2 * 4 + nt, lane), c);
    store_relu(H, c, eb1[nt * 16 + n], lane);
  }
  v16h a0 = loadA64(H, 0, lane), a1 = loadA64(H, 1, lane);
#pragma unroll
  for (int nt = 0; nt < 4; ++nt) {
    v8f c = zero8f();
    c = wmma16(a0, loadB(eW2, nt, lane), c);
    c = wmma16(a1, loadB(eW2, 4 + nt, lane), c);
    store_relu(H, c, eb2[nt * 16 + n], lane);
  }
  a0 = loadA64(H, 0, lane); a1 = loadA64(H, 1, lane);
  v8f c = zero8f();
  c = wmma16(a0, loadB(eW3, 0, lane), c);
  c = wmma16(a1, loadB(eW3, 1, lane), c);
  c = wmma16(A0, loadB(eWr, 0, lane), c);
  c = wmma16(A1, loadB(eWr, 1, lane), c);
  c = wmma16(A2, loadB(eWr, 2, lane), c);
  const float cb = eb3[n] + ebr[n];
#pragma unroll
  for (int v = 0; v < 8; ++v)
    x0g[(size_t)(b0 + v + s8) * 16 + n] = (_Float16)(c[v] + cb);
}

// ---------------------------------------------------------------------------
// Fused recurrence + output map. One wave per 16-row batch tile, 4 waves/block.
// Per step: y_t = h(x_t) (15 WMMAs) and x_{t+1} = f([x_t,u_t]) (15 WMMAs) —
// two independent chains per iteration for latency hiding.
// LDS-relative tiles: fW1 @0(4) fW2 @4(8) fW3 @12(2) fWr @14(1)
//                     hW1 @15(4) hW2 @19(8) hW3 @27(2) hWr @29(1)
// ---------------------------------------------------------------------------
__global__ __launch_bounds__(128) void recur_kernel(
    const _Float16* __restrict__ wpack,   // global packed base (f starts @tile 25)
    const _Float16* __restrict__ x0g,     // [B][16] f16
    const float* __restrict__ ufut,       // [B][T][2] f32
    const float* __restrict__ fb1, const float* __restrict__ fb2,
    const float* __restrict__ fb3, const float* __restrict__ fbr,
    const float* __restrict__ hb1, const float* __restrict__ hb2,
    const float* __restrict__ hb3, const float* __restrict__ hbr,
    float* __restrict__ yout) {           // [B][T][2] f32
  __shared__ __align__(32) _Float16 wlds[30 * 512];
  __shared__ __align__(16) _Float16 hbF[4][16 * 64];
  __shared__ __align__(16) _Float16 hbH[4][16 * 64];
  __shared__ __align__(16) _Float16 xb[4][16 * 16];

  const int tid = threadIdx.x, wave = tid >> 5, lane = tid & 31;

  // copy 30 weight tiles (f+h) to LDS once
  {
    const uint4* src = (const uint4*)(wpack + 25 * 512);
    uint4* dst = (uint4*)wlds;
    for (int i = tid; i < 30 * 512 / 8; i += 128) dst[i] = src[i];
  }
  __syncthreads();

  const int b0 = (blockIdx.x * 4 + wave) * 16;
  const int n = lane & 15, s8 = (lane >> 4) << 3;

  float fB1[4], fB2[4], hB1[4], hB2[4];
#pragma unroll
  for (int j = 0; j < 4; ++j) {
    fB1[j] = fb1[j * 16 + n]; fB2[j] = fb2[j * 16 + n];
    hB1[j] = hb1[j * 16 + n]; hB2[j] = hb2[j * 16 + n];
  }
  const float fCb = fb3[n] + fbr[n];
  const float hCb = (n < 2) ? (hb3[n] + hbr[n]) : 0.f;

  // load x0 tile (16x16 f16, row-major)
  {
    const uint4* src = (const uint4*)(x0g + (size_t)b0 * 16);
    ((uint4*)xb[wave])[lane] = src[lane];
  }

  _Float16* Hf = hbF[wave];
  _Float16* Hh = hbH[wave];
  _Float16* X  = xb[wave];

  const _Float16* fW1 = wlds + 0  * 512;
  const _Float16* fW2 = wlds + 4  * 512;
  const _Float16* fW3 = wlds + 12 * 512;
  const _Float16* fWr = wlds + 14 * 512;
  const _Float16* hW1 = wlds + 15 * 512;
  const _Float16* hW2 = wlds + 19 * 512;
  const _Float16* hW3 = wlds + 27 * 512;
  const _Float16* hWr = wlds + 29 * 512;

#pragma unroll 1
  for (int t = 0; t < TT; ++t) {
    // --- A operands from x_t (16x16 -> K pad 32; u at K=16,17 for f chain) ---
    v8h xlo = *(const v8h*)(X + n * 16 + s8);
    const float2 uv = *(const float2*)(ufut + ((size_t)(b0 + n) * TT + t) * 2);
    v8h utail = zero8h();
    utail[0] = (lane < 16) ? (_Float16)uv.x : (_Float16)0.f;
    utail[1] = (lane < 16) ? (_Float16)uv.y : (_Float16)0.f;
    v16h aH = shuf16(xlo, zero8h());
    v16h aF = shuf16(xlo, utail);

    // ================= h chain: y_t = h(x_t) =================
#pragma unroll
    for (int nt = 0; nt < 4; ++nt) {
      v8f c = zero8f();
      c = wmma16(aH, loadB(hW1, nt, lane), c);
      store_relu(Hh, c, hB1[nt], lane);
    }
    {
      v16h a0 = loadA64(Hh, 0, lane), a1 = loadA64(Hh, 1, lane);
#pragma unroll
      for (int nt = 0; nt < 4; ++nt) {
        v8f c = zero8f();
        c = wmma16(a0, loadB(hW2, nt, lane), c);
        c = wmma16(a1, loadB(hW2, 4 + nt, lane), c);
        store_relu(Hh, c, hB2[nt], lane);
      }
      a0 = loadA64(Hh, 0, lane); a1 = loadA64(Hh, 1, lane);
      v8f cy = zero8f();
      cy = wmma16(a0, loadB(hW3, 0, lane), cy);
      cy = wmma16(a1, loadB(hW3, 1, lane), cy);
      cy = wmma16(aH, loadB(hWr, 0, lane), cy);
      if (n < 2) {
#pragma unroll
        for (int v = 0; v < 8; ++v)
          yout[((size_t)(b0 + v + s8) * TT + t) * 2 + n] = cy[v] + hCb;
      }
    }

    // ================= f chain: x_{t+1} = f([x_t, u_t]) =================
#pragma unroll
    for (int nt = 0; nt < 4; ++nt) {
      v8f c = zero8f();
      c = wmma16(aF, loadB(fW1, nt, lane), c);
      store_relu(Hf, c, fB1[nt], lane);
    }
    {
      v16h a0 = loadA64(Hf, 0, lane), a1 = loadA64(Hf, 1, lane);
#pragma unroll
      for (int nt = 0; nt < 4; ++nt) {
        v8f c = zero8f();
        c = wmma16(a0, loadB(fW2, nt, lane), c);
        c = wmma16(a1, loadB(fW2, 4 + nt, lane), c);
        store_relu(Hf, c, fB2[nt], lane);
      }
      a0 = loadA64(Hf, 0, lane); a1 = loadA64(Hf, 1, lane);
      v8f cx = zero8f();
      cx = wmma16(a0, loadB(fW3, 0, lane), cx);
      cx = wmma16(a1, loadB(fW3, 1, lane), cx);
      cx = wmma16(aF, loadB(fWr, 0, lane), cx);
#pragma unroll
      for (int v = 0; v < 8; ++v)
        X[(v + s8) * 16 + n] = (_Float16)(cx[v] + fCb);   // no relu on state
    }
  }
}

// ---------------------------------------------------------------------------
extern "C" void kernel_launch(void* const* d_in, const int* in_sizes, int n_in,
                              void* d_out, int out_size, void* d_ws, size_t ws_size,
                              hipStream_t stream) {
  (void)in_sizes; (void)n_in; (void)out_size; (void)ws_size;
  const float* upast = (const float*)d_in[0];
  const float* ypast = (const float*)d_in[1];
  const float* ufut  = (const float*)d_in[2];
  const float *eWr = (const float*)d_in[3],  *ebr = (const float*)d_in[4],
              *eW1 = (const float*)d_in[5],  *eb1 = (const float*)d_in[6],
              *eW2 = (const float*)d_in[7],  *eb2 = (const float*)d_in[8],
              *eW3 = (const float*)d_in[9],  *eb3 = (const float*)d_in[10];
  const float *fWr = (const float*)d_in[11], *fbr = (const float*)d_in[12],
              *fW1 = (const float*)d_in[13], *fb1 = (const float*)d_in[14],
              *fW2 = (const float*)d_in[15], *fb2 = (const float*)d_in[16],
              *fW3 = (const float*)d_in[17], *fb3 = (const float*)d_in[18];
  const float *hWr = (const float*)d_in[19], *hbr = (const float*)d_in[20],
              *hW1 = (const float*)d_in[21], *hb1 = (const float*)d_in[22],
              *hW2 = (const float*)d_in[23], *hb2 = (const float*)d_in[24],
              *hW3 = (const float*)d_in[25], *hb3 = (const float*)d_in[26];

  _Float16* wpack = (_Float16*)d_ws;                       // 55 tiles * 1KB
  _Float16* x0g   = (_Float16*)((char*)d_ws + 55 * 1024);  // [B][16] f16

  // pack all weights into WMMA-B layout (tile offsets in halfs = tile*512)
  pack_kernel<<<12, 512, 0, stream>>>(eW1, wpack + 0  * 512, 80, 64, 4);
  pack_kernel<<< 8, 512, 0, stream>>>(eW2, wpack + 12 * 512, 64, 64, 4);
  pack_kernel<<< 2, 512, 0, stream>>>(eW3, wpack + 20 * 512, 64, 16, 1);
  pack_kernel<<< 3, 512, 0, stream>>>(eWr, wpack + 22 * 512, 80, 16, 1);
  pack_kernel<<< 4, 512, 0, stream>>>(fW1, wpack + 25 * 512, 18, 64, 4);
  pack_kernel<<< 8, 512, 0, stream>>>(fW2, wpack + 29 * 512, 64, 64, 4);
  pack_kernel<<< 2, 512, 0, stream>>>(fW3, wpack + 37 * 512, 64, 16, 1);
  pack_kernel<<< 1, 512, 0, stream>>>(fWr, wpack + 39 * 512, 18, 16, 1);
  pack_kernel<<< 4, 512, 0, stream>>>(hW1, wpack + 40 * 512, 16, 64, 4);
  pack_kernel<<< 8, 512, 0, stream>>>(hW2, wpack + 44 * 512, 64, 64, 4);
  pack_kernel<<< 2, 512, 0, stream>>>(hW3, wpack + 52 * 512, 64,  2, 1);
  pack_kernel<<< 1, 512, 0, stream>>>(hWr, wpack + 54 * 512, 16,  2, 1);

  // x0 for all batch rows (512 tiles, 4 waves/block)
  encoder_kernel<<<BATCH / 64, 128, 0, stream>>>(upast, ypast, wpack,
                                                 eb1, eb2, eb3, ebr, x0g);
  // fused T-loop: recurrence + output map
  recur_kernel<<<BATCH / 64, 128, 0, stream>>>(wpack, x0g, ufut,
                                               fb1, fb2, fb3, fbr,
                                               hb1, hb2, hb3, hbr,
                                               (float*)d_out);
}
